// MyRNN_56856777065103
// MI455X (gfx1250) — matrix operand order repaired and checked by reading, not compiled, not run on hardware
//
#include <hip/hip_runtime.h>
#include <hip/hip_bf16.h>

// ---------------------------------------------------------------------------
// Fused 2-layer LSTM + dense head for gfx1250 (MI455X), bf16 WMMA, f32 accum.
//   tokens[B,T] -> emb gather -> LSTM1(seq) -> LSTM2(last) -> sigmoid(h@Wd+bd)
// One workgroup owns a 16-row batch tile for all 80 timesteps. Weight (B)
// WMMA fragments are loop-invariant per wave and live entirely in VGPRs
// (10 x v16bf = 80 VGPRs/lane); only activation (A) fragments stream from
// LDS. h/c state persistent in LDS/registers. Activations use the native
// V_TANH_F32 transcendental (sigmoid via tanh identity) — no IEEE division.
// ---------------------------------------------------------------------------

typedef __attribute__((ext_vector_type(16))) __bf16 v16bf;
typedef __attribute__((ext_vector_type(8)))  float  v8f;

namespace {
constexpr int SEQ    = 80;
constexpr int EMB    = 100;
constexpr int EMBP   = 128;   // EMB padded to K-multiple of 32
constexpr int UNITS  = 64;
constexpr int NGATE  = 256;   // 4*UNITS
constexpr int BM     = 16;    // batch rows per workgroup
constexpr int NTHR   = 512;   // 16 waves of 32
// LDS partition (bytes) — weights now live in VGPRs, not LDS
constexpr int Z_OFF   = 0;                          // f32 [16][256]
constexpr int Z_BYTES = BM * NGATE * 4;             // 16384
constexpr int X_OFF   = Z_OFF + Z_BYTES;            // bf16 [2][16][128] dbl-buf
constexpr int X_BY    = 2 * BM * EMBP * 2;          // 8192
constexpr int H1_OFF  = X_OFF + X_BY;               // bf16 [16][64]
constexpr int H1_BY   = BM * UNITS * 2;             // 2048
constexpr int H2_OFF  = H1_OFF + H1_BY;
constexpr int H2_BY   = H1_BY;
constexpr int TOK_OFF = H2_OFF + H2_BY;             // int [16][80]
constexpr int TOK_BY  = BM * SEQ * 4;               // 5120
constexpr int SMEM_BYTES = TOK_OFF + TOK_BY;        // 33792 bytes
}

union Frag16 { v16bf v; uint4 q[2]; __bf16 e[16]; };

// A-fragment (16x32 bf16, MxK) from row-major LDS tile, per ISA 7.12.2:
//   lanes 0-15: M=lane,    K = k0 + {0..7, 16..23}
//   lanes16-31: M=lane-16, K = k0 + {8..15, 24..31}
__device__ inline v16bf load_a_frag(const __bf16* buf, int stride, int lane, int k0) {
    const __bf16* p = buf + (lane & 15) * stride + k0 + ((lane >> 4) << 3);
    Frag16 f;
    f.q[0] = *(const uint4*)(p);
    f.q[1] = *(const uint4*)(p + 16);
    return f.v;
}

// Build this wave's loop-invariant B-fragment (32x16 bf16, KxN) straight from
// the global f32 weight [Kreal][256], zero-padded in K. Per ISA 7.12.2:
//   lanes 0-15: N=lane,    K = kstep*32 + {0..7, 16..23}
//   lanes16-31: N=lane-16, K = kstep*32 + {8..15, 24..31}
__device__ inline v16bf make_b_frag(const float* __restrict__ g, int Kreal,
                                    int nt, int kstep, int lane) {
    const int N     = nt * 16 + (lane & 15);
    const int Kbase = kstep * 32 + ((lane >> 4) << 3);
    Frag16 f;
#pragma unroll
    for (int e = 0; e < 16; ++e) {
        int K = Kbase + (e < 8 ? e : e + 8);
        f.e[e] = (__bf16)((K < Kreal) ? g[K * NGATE + N] : 0.f);
    }
    return f.v;
}

// Native transcendental tanh (V_TANH_F32 on gfx1250); fallback avoids the
// IEEE-divide expansion by using the fast v_rcp_f32.
__device__ inline float fast_tanh(float x) {
#if __has_builtin(__builtin_amdgcn_tanhf)
    return __builtin_amdgcn_tanhf(x);
#else
    float e = __expf(2.f * x);
    return 1.f - 2.f * __builtin_amdgcn_rcpf(e + 1.f);
#endif
}
__device__ inline float fast_sigmoid(float x) {
    return fmaf(0.5f, fast_tanh(0.5f * x), 0.5f);
}

__global__ __launch_bounds__(NTHR)
void lstm2_fused_kernel(const int* __restrict__ tokens,
                        const float* __restrict__ emb,
                        const float* __restrict__ W1, const float* __restrict__ U1,
                        const float* __restrict__ b1,
                        const float* __restrict__ W2, const float* __restrict__ U2,
                        const float* __restrict__ b2,
                        const float* __restrict__ Wd, const float* __restrict__ bd,
                        float* __restrict__ out) {
    extern __shared__ char smem[];
    float*  zbuf  = (float*)(smem + Z_OFF);    // [16][256] gate pre/post-act
    __bf16* xbuf  = (__bf16*)(smem + X_OFF);   // [2][16][128]
    __bf16* h1buf = (__bf16*)(smem + H1_OFF);  // [16][64]
    __bf16* h2buf = (__bf16*)(smem + H2_OFF);  // [16][64]
    int*    tokb  = (int*)(smem + TOK_OFF);    // [16][80]

    const int tid  = threadIdx.x;
    const int lane = tid & 31;
    const int nt   = tid >> 5;          // wave id == N-tile (gate column tile)
    const int b0   = blockIdx.x * BM;   // first batch row of this tile

    // --- one-time: B-fragments -> registers, tokens -> LDS, h0 = 0 ---------
    v16bf w1r[4], u1r[2], w2r[2], u2r[2];
#pragma unroll
    for (int ks = 0; ks < 4; ++ks) w1r[ks] = make_b_frag(W1, EMB,   nt, ks, lane);
#pragma unroll
    for (int ks = 0; ks < 2; ++ks) u1r[ks] = make_b_frag(U1, UNITS, nt, ks, lane);
#pragma unroll
    for (int ks = 0; ks < 2; ++ks) w2r[ks] = make_b_frag(W2, UNITS, nt, ks, lane);
#pragma unroll
    for (int ks = 0; ks < 2; ++ks) u2r[ks] = make_b_frag(U2, UNITS, nt, ks, lane);

    for (int i = tid; i < BM * UNITS; i += NTHR) {
        h1buf[i] = (__bf16)0.f;
        h2buf[i] = (__bf16)0.f;
    }
    for (int i = tid; i < BM * SEQ; i += NTHR)
        tokb[i] = tokens[(b0 + i / SEQ) * SEQ + (i % SEQ)];
    __syncthreads();  // tokb valid before gather below

    // gather x for t=0 into buffer 0 (f32 -> bf16, K padded to 128)
    for (int idx = tid; idx < BM * EMBP; idx += NTHR) {
        int m = idx >> 7, k = idx & (EMBP - 1);
        int tok = tokb[m * SEQ];
        xbuf[m * EMBP + k] = (__bf16)((k < EMB) ? emb[tok * EMB + k] : 0.f);
    }

    // per-wave constants
    const int  ncol    = nt * 16 + (lane & 15);  // global gate column
    const int  half    = lane >> 4;
    const bool is_tanh = ((nt >> 2) == 2);       // gate order i,f,g,o; g=tanh
    // act(v) = s*tanh(s*v) + o : (s,o) = (1,0) for tanh, (0.5,0.5) for sigmoid
    const float act_s  = is_tanh ? 1.f : 0.5f;
    const float act_o  = is_tanh ? 0.f : 0.5f;
    const float bias1  = b1[ncol];
    const float bias2  = b2[ncol];

    float c1[2] = {0.f, 0.f};
    float c2[2] = {0.f, 0.f};

    __syncthreads();

    for (int t = 0; t < SEQ; ++t) {
        const __bf16* xcur = xbuf + (t & 1) * BM * EMBP;
        __bf16*       xnxt = xbuf + ((t + 1) & 1) * BM * EMBP;

        // ---- Phase B: layer-1 gates  z1 = x@W1 + h1@U1 + b1, activated
        {
            v8f acc = {};
#pragma unroll
            for (int ks = 0; ks < 4; ++ks) {   // K over padded EMB (128)
                v16bf a = load_a_frag(xcur, EMBP, lane, ks * 32);
                acc = __builtin_amdgcn_wmma_f32_16x16x32_bf16(
                    false, a, false, w1r[ks], (short)0, acc, false, false);
            }
#pragma unroll
            for (int ks = 0; ks < 2; ++ks) {   // K over UNITS (64)
                v16bf a = load_a_frag(h1buf, UNITS, lane, ks * 32);
                acc = __builtin_amdgcn_wmma_f32_16x16x32_bf16(
                    false, a, false, u1r[ks], (short)0, acc, false, false);
            }
            // C/D layout: VGPR r -> M = r + 8*half, N = lane&15
#pragma unroll
            for (int r = 0; r < 8; ++r) {
                float v = acc[r] + bias1;
                zbuf[(r + 8 * half) * NGATE + ncol] =
                    fmaf(act_s, fast_tanh(act_s * v), act_o);
            }
        }
        __syncthreads();

        // ---- Phase C: layer-1 state update, h1 -> LDS (bf16)
#pragma unroll
        for (int j = 0; j < 2; ++j) {
            int p = tid + j * NTHR;            // (m,u) pair
            int m = p >> 6, u = p & 63;
            float ig = zbuf[m * NGATE + u];
            float fg = zbuf[m * NGATE + 64 + u];
            float gg = zbuf[m * NGATE + 128 + u];
            float og = zbuf[m * NGATE + 192 + u];
            float c  = fg * c1[j] + ig * gg;
            c1[j] = c;
            h1buf[m * UNITS + u] = (__bf16)(og * fast_tanh(c));
        }
        __syncthreads();

        // ---- Phase D: layer-2 gates  z2 = h1@W2 + h2@U2 + b2, activated
        {
            v8f acc = {};
#pragma unroll
            for (int ks = 0; ks < 2; ++ks) {
                v16bf a = load_a_frag(h1buf, UNITS, lane, ks * 32);
                acc = __builtin_amdgcn_wmma_f32_16x16x32_bf16(
                    false, a, false, w2r[ks], (short)0, acc, false, false);
            }
#pragma unroll
            for (int ks = 0; ks < 2; ++ks) {
                v16bf a = load_a_frag(h2buf, UNITS, lane, ks * 32);
                acc = __builtin_amdgcn_wmma_f32_16x16x32_bf16(
                    false, a, false, u2r[ks], (short)0, acc, false, false);
            }
#pragma unroll
            for (int r = 0; r < 8; ++r) {
                float v = acc[r] + bias2;
                zbuf[(r + 8 * half) * NGATE + ncol] =
                    fmaf(act_s, fast_tanh(act_s * v), act_o);
            }
        }
        __syncthreads();

        // ---- Phase E: layer-2 state update; gather x_{t+1}; stash f32 h2
#pragma unroll
        for (int j = 0; j < 2; ++j) {
            int p = tid + j * NTHR;
            int m = p >> 6, u = p & 63;
            float ig = zbuf[m * NGATE + u];
            float fg = zbuf[m * NGATE + 64 + u];
            float gg = zbuf[m * NGATE + 128 + u];
            float og = zbuf[m * NGATE + 192 + u];
            float c  = fg * c2[j] + ig * gg;
            c2[j] = c;
            float h = og * fast_tanh(c);
            h2buf[m * UNITS + u] = (__bf16)h;
            if (t == SEQ - 1)                    // own slot, safe post-read
                zbuf[m * NGATE + u] = h;         // keep full-precision h2
        }
        if (t + 1 < SEQ) {
            // gather next step's x into the other buffer (hidden behind E)
            for (int idx = tid; idx < BM * EMBP; idx += NTHR) {
                int m = idx >> 7, k = idx & (EMBP - 1);
                int tok = tokb[m * SEQ + t + 1];
                xnxt[m * EMBP + k] =
                    (__bf16)((k < EMB) ? emb[tok * EMB + k] : 0.f);
            }
            if (t + 2 < SEQ && tid < BM) {       // warm t+2 rows (prefetch_b8)
                int tok2 = tokb[tid * SEQ + t + 2];
                __builtin_prefetch(&emb[tok2 * EMB], 0, 1);
            }
        }
        __syncthreads();
    }

    // ---- Dense head: sigmoid(h2 @ Wd + bd), one row per thread
    if (tid < BM) {
        float acc = bd[0];
        for (int u = 0; u < UNITS; ++u)
            acc += zbuf[tid * NGATE + u] * Wd[u];
        out[b0 + tid] = fast_sigmoid(acc);
    }
}

extern "C" void kernel_launch(void* const* d_in, const int* in_sizes, int n_in,
                              void* d_out, int out_size, void* d_ws, size_t ws_size,
                              hipStream_t stream) {
    (void)in_sizes; (void)n_in; (void)out_size; (void)d_ws; (void)ws_size;
    const int*   tokens = (const int*)d_in[0];
    const float* emb    = (const float*)d_in[1];
    const float* W1     = (const float*)d_in[2];
    const float* U1     = (const float*)d_in[3];
    const float* b1     = (const float*)d_in[4];
    const float* W2     = (const float*)d_in[5];
    const float* U2     = (const float*)d_in[6];
    const float* b2     = (const float*)d_in[7];
    const float* Wd     = (const float*)d_in[8];
    const float* bd     = (const float*)d_in[9];
    float* out = (float*)d_out;

    (void)hipFuncSetAttribute((const void*)lstm2_fused_kernel,
                              hipFuncAttributeMaxDynamicSharedMemorySize,
                              SMEM_BYTES);
    const int BATCH = 4096;
    lstm2_fused_kernel<<<BATCH / BM, NTHR, SMEM_BYTES, stream>>>(
        tokens, emb, W1, U1, b1, W2, U2, b2, Wd, bd, out);
}
